// SimCLR_26920855011622
// MI455X (gfx1250) — matrix operand order repaired and checked by reading, not compiled, not run on hardware
//
#include <hip/hip_runtime.h>

typedef __attribute__((ext_vector_type(16))) _Float16 v16h;
typedef __attribute__((ext_vector_type(8)))  float    v8f;

#define B_    4096
#define DIN   192
#define DHID  512
#define DOUT  128
#define NROWS 8192          // 2 views concatenated
#define INV_T 2.0f          // 1 / 0.5

__device__ __forceinline__ v8f wmma_f16(v16h a, v16h b, v8f c) {
  return __builtin_amdgcn_wmma_f32_16x16x32_f16(false, a, false, b, (short)0, c, false, false);
}

// ---------------------------------------------------------------------------
// K0: one-time f32 -> f16 conversion (h1,h2,W1,W2). 4 elems/thread.
// ---------------------------------------------------------------------------
__global__ __launch_bounds__(256) void k_cvt(const float* __restrict__ src,
                                             _Float16* __restrict__ dst, int n) {
  const int i = (blockIdx.x * 256 + threadIdx.x) * 4;
  if (i + 3 < n) {
    const float4 v = *(const float4*)(src + i);
    dst[i + 0] = (_Float16)v.x;
    dst[i + 1] = (_Float16)v.y;
    dst[i + 2] = (_Float16)v.z;
    dst[i + 3] = (_Float16)v.w;
  }
}

// ---------------------------------------------------------------------------
// K1: Y[R, c] = [h1;h2][R,:] @ W1[c,:] + b1[c]   (8192 x 512, K=192)
// One wave per 16x64 output tile. f16 operands pre-converted; pure WMMA loop.
// ---------------------------------------------------------------------------
__global__ __launch_bounds__(256) void k_gemm1(const _Float16* __restrict__ Hh,
                                               const _Float16* __restrict__ W1h,
                                               const float* __restrict__ b1,
                                               float* __restrict__ Y) {
  const int lane = threadIdx.x & 31;
  const int wave = threadIdx.x >> 5;
  const int task = blockIdx.x * 8 + wave;   // 4096 tiles
  const int rt = task >> 3;                 // 512 row tiles
  const int ct = task & 7;                  // 8 col tiles of 64
  const int R0 = rt << 4;
  const int C0 = ct << 6;
  const int n  = lane & 15;                 // A row m == B col n == lane&15
  const int hi = lane >> 4;

  const _Float16* arow = Hh + (size_t)(R0 + n) * DIN;

  v8f acc[4] = {};
  for (int kk = 0; kk < DIN; kk += 32) {
    v16h a;
#pragma unroll
    for (int e = 0; e < 8; ++e) {
      a[e]     = arow[kk + 16 * hi + e];       // halves 0-7 : K = kk+16hi+e
      a[8 + e] = arow[kk + 16 * hi + 8 + e];   // halves 8-15: K = kk+16hi+8+e
    }
#pragma unroll
    for (int t = 0; t < 4; ++t) {
      const int c = C0 + t * 16 + n;
      const v16h b = *(const v16h*)(W1h + (size_t)c * DIN + kk + 16 * hi);
      acc[t] = wmma_f16(a, b, acc[t]);
    }
  }
#pragma unroll
  for (int t = 0; t < 4; ++t) {
    const int c = C0 + t * 16 + n;
    const float bias = b1[c];
#pragma unroll
    for (int v = 0; v < 8; ++v) {
      const int R = R0 + v + 8 * hi;           // D: vgpr v -> M = v + 8*hi
      Y[(size_t)R * DHID + c] = acc[t][v] + bias;
    }
  }
}

// ---------------------------------------------------------------------------
// K2: per-view, per-channel BN stats folded to scale/shift.
// grid = 64 blocks = (view, 16-col group); lanes read consecutive cols.
// ---------------------------------------------------------------------------
__global__ __launch_bounds__(256) void k_bnstats(const float* __restrict__ Y,
                                                 const float* __restrict__ gamma,
                                                 const float* __restrict__ beta,
                                                 float* __restrict__ scaleA,
                                                 float* __restrict__ shiftA) {
  const int view = blockIdx.x >> 5;
  const int col0 = (blockIdx.x & 31) << 4;
  const int c  = threadIdx.x & 15;   // column within group
  const int rg = threadIdx.x >> 4;   // 16 row groups
  const float* base = Y + (size_t)view * B_ * DHID + col0 + c;
  float s = 0.f, s2 = 0.f;
  for (int r = rg; r < B_; r += 16) {
    const float v = base[(size_t)r * DHID];
    s += v; s2 += v * v;
  }
  __shared__ float ls[16][17], ls2[16][17];
  ls[rg][c] = s; ls2[rg][c] = s2;
  __syncthreads();
  if (threadIdx.x < 16) {
    float ts = 0.f, ts2 = 0.f;
    for (int g = 0; g < 16; ++g) { ts += ls[g][threadIdx.x]; ts2 += ls2[g][threadIdx.x]; }
    const int col = col0 + threadIdx.x;
    const float m   = ts * (1.0f / B_);
    const float var = ts2 * (1.0f / B_) - m * m;   // biased var
    const float sc  = rsqrtf(var + 1e-5f) * gamma[col];
    scaleA[view * DHID + col] = sc;
    shiftA[view * DHID + col] = beta[col] - m * sc;
  }
}

// ---------------------------------------------------------------------------
// K3: Z = relu(Y*scale+shift) @ W2^T + b2, then L2-normalize rows.
// One wave per 16-row tile x all 128 cols (8 n-subtiles, K=512).
// ---------------------------------------------------------------------------
__global__ __launch_bounds__(256) void k_bn_gemm2(const float* __restrict__ Y,
                                                  const float* __restrict__ scaleA,
                                                  const float* __restrict__ shiftA,
                                                  const _Float16* __restrict__ W2h,
                                                  const float* __restrict__ b2,
                                                  float* __restrict__ zout,
                                                  _Float16* __restrict__ zh) {
  const int lane = threadIdx.x & 31;
  const int wave = threadIdx.x >> 5;
  const int rt = blockIdx.x * 8 + wave;   // 512 tiles
  const int R0 = rt << 4;
  const int n  = lane & 15;
  const int hi = lane >> 4;
  const int view = (R0 >= B_);
  const float* sc = scaleA + view * DHID;
  const float* sh = shiftA + view * DHID;
  const float* yrow = Y + (size_t)(R0 + n) * DHID;

  v8f acc[8] = {};
  for (int kk = 0; kk < DHID; kk += 32) {
    v16h a;
#pragma unroll
    for (int e = 0; e < 8; ++e) {
      const int k0 = kk + 16 * hi + e;
      const int k1 = kk + 16 * hi + 8 + e;
      const float y0 = fmaf(yrow[k0], sc[k0], sh[k0]);
      const float y1 = fmaf(yrow[k1], sc[k1], sh[k1]);
      a[e]     = (_Float16)fmaxf(y0, 0.f);
      a[8 + e] = (_Float16)fmaxf(y1, 0.f);
    }
#pragma unroll
    for (int t = 0; t < 8; ++t) {
      const int c = t * 16 + n;
      const v16h b = *(const v16h*)(W2h + (size_t)c * DHID + kk + 16 * hi);
      acc[t] = wmma_f16(a, b, acc[t]);
    }
  }
#pragma unroll
  for (int t = 0; t < 8; ++t) {
    const float bias = b2[t * 16 + n];
#pragma unroll
    for (int v = 0; v < 8; ++v) acc[t][v] += bias;
  }
#pragma unroll
  for (int v = 0; v < 8; ++v) {
    float ss = 0.f;
#pragma unroll
    for (int t = 0; t < 8; ++t) ss += acc[t][v] * acc[t][v];
    for (int off = 1; off < 16; off <<= 1) ss += __shfl_xor(ss, off, 16);
    const float inv = 1.0f / fmaxf(sqrtf(ss), 1e-12f);
    const int R = R0 + v + 8 * hi;
#pragma unroll
    for (int t = 0; t < 8; ++t) {
      const float zv = acc[t][v] * inv;
      const int c = t * 16 + n;
      zout[(size_t)R * DOUT + c] = zv;
      zh[(size_t)R * DOUT + c]  = (_Float16)zv;
    }
  }
}

// ---------------------------------------------------------------------------
// K4: fused sim-GEMM + masked online exp-sum -> rowloss[i] = lse_i - pos_i.
// Block = 16 rows; 8 waves split 512 j-tiles, 2 tiles per iteration so two
// independent WMMA accumulator chains interleave (hides WMMA RAW hazards).
// ---------------------------------------------------------------------------
__global__ __launch_bounds__(256) void k_ntxent(const _Float16* __restrict__ zh,
                                                float* __restrict__ rowloss) {
  const int lane = threadIdx.x & 31;
  const int wave = threadIdx.x >> 5;
  const int R0 = blockIdx.x << 4;
  const int n  = lane & 15;
  const int hi = lane >> 4;

  const _Float16* arow = zh + (size_t)(R0 + n) * DOUT;
  v16h a[4];
#pragma unroll
  for (int kc = 0; kc < 4; ++kc)
#pragma unroll
    for (int e = 0; e < 8; ++e) {
      a[kc][e]     = arow[kc * 32 + 16 * hi + e];
      a[kc][8 + e] = arow[kc * 32 + 16 * hi + 8 + e];
    }

  float esum[8] = {};
  float pos[8]  = {};
  for (int jt = wave * 2; jt < NROWS / 16; jt += 16) {
    const int C0a = jt << 4;
    const int C0b = C0a + 16;
    const _Float16* browA = zh + (size_t)(C0a + n) * DOUT + 16 * hi;
    const _Float16* browB = zh + (size_t)(C0b + n) * DOUT + 16 * hi;
    v8f accA = {}, accB = {};
#pragma unroll
    for (int kc = 0; kc < 4; ++kc) {
      const v16h ba = *(const v16h*)(browA + kc * 32);
      const v16h bb = *(const v16h*)(browB + kc * 32);
      accA = wmma_f16(a[kc], ba, accA);
      accB = wmma_f16(a[kc], bb, accB);
    }
    const int ja = C0a + n;
    const int jb = C0b + n;
#pragma unroll
    for (int v = 0; v < 8; ++v) {
      const int i = R0 + v + 8 * hi;
      const float simA = accA[v] * INV_T;    // |sim| <= 2
      const float simB = accB[v] * INV_T;
      if (ja != i) esum[v] += __expf(simA);  // diagonal masked (== -inf)
      if (jb != i) esum[v] += __expf(simB);
      if ((ja ^ i) == B_) pos[v] = simA;     // positive pair i <-> i±B
      if ((jb ^ i) == B_) pos[v] = simB;
    }
  }

  __shared__ float se[8 * 16], sp[8 * 16];
#pragma unroll
  for (int v = 0; v < 8; ++v) {
    float e = esum[v], p = pos[v];
    for (int off = 1; off < 16; off <<= 1) {
      e += __shfl_xor(e, off, 16);
      p += __shfl_xor(p, off, 16);
    }
    if (n == 0) {
      const int row = v + 8 * hi;
      se[wave * 16 + row] = e;
      sp[wave * 16 + row] = p;
    }
  }
  __syncthreads();
  if (threadIdx.x < 16) {
    float te = 0.f, tp = 0.f;
    for (int w = 0; w < 8; ++w) { te += se[w * 16 + threadIdx.x]; tp += sp[w * 16 + threadIdx.x]; }
    rowloss[R0 + threadIdx.x] = logf(te) - tp;
  }
}

// ---------------------------------------------------------------------------
// K5: loss = mean(rowloss)
// ---------------------------------------------------------------------------
__global__ __launch_bounds__(256) void k_loss(const float* __restrict__ rowloss,
                                              float* __restrict__ out) {
  __shared__ float ls[256];
  float s = 0.f;
  for (int i = threadIdx.x; i < NROWS; i += 256) s += rowloss[i];
  ls[threadIdx.x] = s;
  __syncthreads();
  for (int off = 128; off > 0; off >>= 1) {
    if (threadIdx.x < off) ls[threadIdx.x] += ls[threadIdx.x + off];
    __syncthreads();
  }
  if (threadIdx.x == 0) out[0] = ls[0] * (1.0f / NROWS);
}

extern "C" void kernel_launch(void* const* d_in, const int* in_sizes, int n_in,
                              void* d_out, int out_size, void* d_ws, size_t ws_size,
                              hipStream_t stream) {
  const float* h1    = (const float*)d_in[0];
  const float* h2    = (const float*)d_in[1];
  const float* W1    = (const float*)d_in[2];
  const float* b1    = (const float*)d_in[3];
  const float* gamma = (const float*)d_in[4];
  const float* beta  = (const float*)d_in[5];
  const float* W2    = (const float*)d_in[6];
  const float* b2    = (const float*)d_in[7];
  float* out = (float*)d_out;   // [0]=loss, [1..]=z1 flat, then z2 flat

  float* Y        = (float*)d_ws;                        // 8192*512 f32 (16 MB)
  float* scaleA   = Y + (size_t)NROWS * DHID;            // 1024 f32
  float* shiftA   = scaleA + 2 * DHID;                   // 1024 f32
  float* rowloss  = shiftA + 2 * DHID;                   // 8192 f32
  _Float16* zh    = (_Float16*)(rowloss + NROWS);        // 8192*128 f16 (2 MB)
  _Float16* Hh    = zh + (size_t)NROWS * DOUT;           // 8192*192 f16 (3 MB)
  _Float16* W1h   = Hh + (size_t)NROWS * DIN;            // 512*192 f16
  _Float16* W2h   = W1h + (size_t)DHID * DIN;            // 128*512 f16

  const int nH = B_ * DIN;       // 786432
  k_cvt<<<nH / 1024, 256, 0, stream>>>(h1, Hh, nH);
  k_cvt<<<nH / 1024, 256, 0, stream>>>(h2, Hh + nH, nH);
  k_cvt<<<(DHID * DIN) / 1024, 256, 0, stream>>>(W1, W1h, DHID * DIN);
  k_cvt<<<(DOUT * DHID) / 1024, 256, 0, stream>>>(W2, W2h, DOUT * DHID);

  k_gemm1   <<<512, 256, 0, stream>>>(Hh, W1h, b1, Y);
  k_bnstats <<<64, 256, 0, stream>>>(Y, gamma, beta, scaleA, shiftA);
  k_bn_gemm2<<<64, 256, 0, stream>>>(Y, scaleA, shiftA, W2h, b2, out + 1, zh);
  k_ntxent  <<<512, 256, 0, stream>>>(zh, rowloss);
  k_loss    <<<1, 256, 0, stream>>>(rowloss, out);
}
